// Unet_33243046871542
// MI455X (gfx1250) — compile-verified, matrix operand order checked by reading, not run on hardware
//
#include <hip/hip_runtime.h>
#include <hip/hip_bf16.h>

typedef __attribute__((ext_vector_type(16))) _Float16 v16h;
typedef __attribute__((ext_vector_type(8)))  _Float16 v8h;
typedef __attribute__((ext_vector_type(8)))  float    v8f;

#define LRELU_SLOPE 0.01f
#define BN_EPS 1e-5f

__device__ __forceinline__ v16h cat8(v8h lo, v8h hh) {
  return __builtin_shufflevector(lo, hh, 0, 1, 2, 3, 4, 5, 6, 7,
                                 8, 9, 10, 11, 12, 13, 14, 15);
}

// ---------------------------------------------------------------------------
// Zero the 1-pixel halo ring of a padded NHWC buffer (N, Hp, Wp, C).
// ---------------------------------------------------------------------------
__global__ void k_zero_border(_Float16* __restrict__ buf, int N, int Hp, int Wp, int C) {
  const int nb = 2 * Wp + 2 * (Hp - 2);  // border pixels per image
  long i = (long)blockIdx.x * blockDim.x + threadIdx.x;
  long total = (long)N * nb * C;
  if (i >= total) return;
  int c = (int)(i % C);
  long t = i / C;
  int b = (int)(t % nb);
  int n = (int)(t / nb);
  int y, x;
  if (b < Wp) { y = 0; x = b; }
  else if (b < 2 * Wp) { y = Hp - 1; x = b - Wp; }
  else { int r = b - 2 * Wp; y = 1 + (r >> 1); x = (r & 1) ? (Wp - 1) : 0; }
  buf[(((size_t)n * Hp + y) * Wp + x) * C + c] = (_Float16)0.f;
}

// ---------------------------------------------------------------------------
// Input convert: NCHW f32 (2,3,512,512) -> padded NHWC f16 (2,514,514,3)
// ---------------------------------------------------------------------------
__global__ void k_cvt_in(const float* __restrict__ src, _Float16* __restrict__ dst,
                         int N, int H, int W) {
  long i = (long)blockIdx.x * blockDim.x + threadIdx.x;  // over N*H*W*3
  if (i >= (long)N * H * W * 3) return;
  long pix = i / 3;
  int c = (int)(i - pix * 3);
  int n = (int)(pix / ((long)H * W));
  int hw = (int)(pix - (long)n * H * W);
  int y = hw / W, x = hw - y * W;
  const int Hp = H + 2, Wp = W + 2;
  dst[(((size_t)n * Hp + y + 1) * Wp + x + 1) * 3 + c] =
      (_Float16)src[((size_t)n * 3 + c) * H * W + hw];
}

// ---------------------------------------------------------------------------
// Pack weights (Cout,Cin,3,3) f32 -> WMMA A-fragment f16, tap-major K:
//   k = tap*Cin + ci   (tap = ky*3+kx)
// Packed: [mt][kt][lane][j], 512 halves per 16x32 chunk; lane j-order per
// §7.12.2 (lane<16: K {0..7,16..23}; lane>=16: K {8..15,24..31}).
// ---------------------------------------------------------------------------
__global__ void k_pack_w(const float* __restrict__ w, _Float16* __restrict__ wp,
                         int Cout, int Cin, int K, int nKt, int total) {
  int i = blockIdx.x * blockDim.x + threadIdx.x;
  if (i >= total) return;
  int j     = i & 15;
  int lane  = (i >> 4) & 31;
  int chunk = i >> 9;
  int kt = chunk % nKt;
  int mt = chunk / nKt;
  int hi = lane >> 4;
  int m  = mt * 16 + (lane & 15);
  int k  = kt * 32 + j + (j >= 8 ? 8 : 0) + (hi ? 8 : 0);
  _Float16 v = (_Float16)0.f;
  if (m < Cout && k < K) {
    int tap = k / Cin;
    int ci  = k - tap * Cin;
    v = (_Float16)w[((size_t)m * Cin + ci) * 9 + tap];
  }
  wp[i] = v;
}

// ---------------------------------------------------------------------------
// Fast implicit-GEMM 3x3 conv, padded NHWC f16 input, Cin % 32 == 0.
// Branchless: halo makes every tap in-bounds; P % 128 == 0 so no partial
// waves. Wave = 16 couts x 32 pixels (two 16x16 WMMA tiles sharing A).
//   stride==2 : conv evaluated at even coords (fused _down2)
//   opad      : 1 -> output written with its own 1-px halo
// ---------------------------------------------------------------------------
__global__ void __launch_bounds__(128)
k_conv3x3_nhwc(const _Float16* __restrict__ in, const _Float16* __restrict__ wp,
               const float* __restrict__ bias,
               _Float16* __restrict__ out_h, float* __restrict__ out_f,
               int N, int Cin, int H, int W, int Cout, int Ho, int Wo,
               int stride, int do_lrelu, int opad) {
  const int lane = threadIdx.x & 31;
  const int wave = threadIdx.x >> 5;
  const int hi   = lane >> 4;
  const int lm   = lane & 15;
  const int cpt  = Cin >> 5;  // K-chunks per tap
  const int nKt  = 9 * cpt;
  const int Hp = H + 2, Wp = W + 2;
  const long hw = (long)Ho * Wo;

  // two pixel tiles per wave (exact: P % 128 == 0)
  long p0 = ((long)blockIdx.y * 4 + wave) * 32 + lm;
  long p1 = p0 + 16;
  int n0 = (int)(p0 / hw);
  int r0 = (int)(p0 - n0 * hw);
  int yo0 = r0 / Wo, xo0 = r0 - yo0 * Wo;
  int n1 = (int)(p1 / hw);
  int r1 = (int)(p1 - n1 * hw);
  int yo1 = r1 / Wo, xo1 = r1 - yo1 * Wo;

  const size_t pix0 = ((size_t)n0 * Hp + yo0 * stride) * Wp + xo0 * stride;
  const size_t pix1 = ((size_t)n1 * Hp + yo1 * stride) * Wp + xo1 * stride;
  const int hioff = hi ? 8 : 0;

  const _Float16* wpA = wp + (size_t)blockIdx.x * nKt * 512 + (size_t)lane * 16;

  v8f acc0 = {}, acc1 = {};
#pragma unroll
  for (int tap = 0; tap < 9; ++tap) {
    const int ty = tap / 3, tx = tap - (tap / 3) * 3;
    const _Float16* b0p = in + (pix0 + (size_t)ty * Wp + tx) * Cin + hioff;
    const _Float16* b1p = in + (pix1 + (size_t)ty * Wp + tx) * Cin + hioff;
    const _Float16* wpT = wpA + (size_t)tap * cpt * 512;
    for (int c = 0; c < cpt; ++c) {
      v16h a = *(const v16h*)(wpT + (size_t)c * 512);
      const int ci0 = c * 32;
      v16h b0 = cat8(*(const v8h*)(b0p + ci0), *(const v8h*)(b0p + ci0 + 16));
      v16h b1 = cat8(*(const v8h*)(b1p + ci0), *(const v8h*)(b1p + ci0 + 16));
      acc0 = __builtin_amdgcn_wmma_f32_16x16x32_f16(false, a, false, b0,
                                                    (short)0, acc0, false, false);
      acc1 = __builtin_amdgcn_wmma_f32_16x16x32_f16(false, a, false, b1,
                                                    (short)0, acc1, false, false);
    }
  }

  const int m0 = blockIdx.x * 16 + hioff;
  const int oHp = Ho + 2 * opad, oWp = Wo + 2 * opad;
  {
    size_t base = (((size_t)n0 * oHp + yo0 + opad) * oWp + xo0 + opad) * Cout + m0;
    v8h oh;
#pragma unroll
    for (int r = 0; r < 8; ++r) {
      float v = acc0[r] + bias[m0 + r];
      if (do_lrelu) v = (v >= 0.f) ? v : LRELU_SLOPE * v;
      oh[r] = (_Float16)v;
      if (out_f) out_f[base + r] = v;
    }
    if (out_h) *(v8h*)(out_h + base) = oh;
  }
  {
    size_t base = (((size_t)n1 * oHp + yo1 + opad) * oWp + xo1 + opad) * Cout + m0;
    v8h oh;
#pragma unroll
    for (int r = 0; r < 8; ++r) {
      float v = acc1[r] + bias[m0 + r];
      if (do_lrelu) v = (v >= 0.f) ? v : LRELU_SLOPE * v;
      oh[r] = (_Float16)v;
      if (out_f) out_f[base + r] = v;
    }
    if (out_h) *(v8h*)(out_h + base) = oh;
  }
}

// ---------------------------------------------------------------------------
// Small-Cin conv (layer 0, Cin=3, K=27): per-element gather, padded NHWC.
// ---------------------------------------------------------------------------
__global__ void __launch_bounds__(128)
k_conv3x3_nhwc_small(const _Float16* __restrict__ in, const _Float16* __restrict__ wp,
                     const float* __restrict__ bias,
                     _Float16* __restrict__ out_h,
                     int N, int Cin, int H, int W, int Cout, int Ho, int Wo,
                     int stride, int do_lrelu, int opad) {
  const int lane = threadIdx.x & 31;
  const int wave = threadIdx.x >> 5;
  const int hi   = lane >> 4;
  const int lm   = lane & 15;
  const int K    = Cin * 9;
  const int nKt  = (K + 31) >> 5;
  const int Hp = H + 2, Wp = W + 2;
  const long hw = (long)Ho * Wo;

  const long p = ((long)blockIdx.y * 4 + wave) * 16 + lm;  // P % 64 == 0
  int n = (int)(p / hw);
  int r = (int)(p - n * hw);
  int yo = r / Wo, xo = r - yo * Wo;
  const size_t pix = ((size_t)n * Hp + yo * stride) * Wp + xo * stride;
  const _Float16* wpA = wp + (size_t)blockIdx.x * nKt * 512 + (size_t)lane * 16;

  v8f acc = {};
  for (int kt = 0; kt < nKt; ++kt) {
    v16h a = *(const v16h*)(wpA + (size_t)kt * 512);
    v16h b;
    const int kc = kt * 32 + (hi ? 8 : 0);
#pragma unroll
    for (int j = 0; j < 16; ++j) {
      const int k = kc + j + (j >= 8 ? 8 : 0);
      _Float16 v = (_Float16)0.f;
      if (k < K) {
        int tap = k / Cin;
        int ci  = k - tap * Cin;
        int ty  = tap / 3;
        int tx  = tap - ty * 3;
        v = in[(pix + (size_t)ty * Wp + tx) * Cin + ci];
      }
      b[j] = v;
    }
    acc = __builtin_amdgcn_wmma_f32_16x16x32_f16(false, a, false, b,
                                                 (short)0, acc, false, false);
  }

  const int m0 = blockIdx.x * 16 + (hi ? 8 : 0);
  const int oHp = Ho + 2 * opad, oWp = Wo + 2 * opad;
  size_t base = (((size_t)n * oHp + yo + opad) * oWp + xo + opad) * Cout + m0;
  v8h oh;
#pragma unroll
  for (int r2 = 0; r2 < 8; ++r2) {
    float v = acc[r2] + bias[m0 + r2];
    if (do_lrelu) v = (v >= 0.f) ? v : LRELU_SLOPE * v;
    oh[r2] = (_Float16)v;
  }
  *(v8h*)(out_h + base) = oh;
}

// ---------------------------------------------------------------------------
// Zero-insert 2x upsample: unpadded NHWC in (N,H,W,C) -> padded (N,2H+2,2W+2,C)
// interior; halo zeroed separately.
// ---------------------------------------------------------------------------
__global__ void k_up2(const _Float16* __restrict__ in, _Float16* __restrict__ out,
                      int N, int C, int H, int W) {
  long i = (long)blockIdx.x * blockDim.x + threadIdx.x;  // over N*2H*2W*C
  long total = (long)N * 4 * H * W * C;
  if (i >= total) return;
  int c = (int)(i % C);
  long t = i / C;
  int W2 = 2 * W;
  int x = (int)(t % W2);
  long t2 = t / W2;
  int y = (int)(t2 % (2 * H));
  int n = (int)(t2 / (2 * H));
  _Float16 v = (_Float16)0.f;
  if (!(y & 1) && !(x & 1))
    v = in[(((size_t)n * H + (y >> 1)) * W + (x >> 1)) * C + c];
  const int Hp = 2 * H + 2, Wp = 2 * W + 2;
  out[(((size_t)n * Hp + y + 1) * Wp + x + 1) * C + c] = v;
}

// ---------------------------------------------------------------------------
// BatchNorm stats over unpadded NHWC f32 (P pixels): one 1024-thread block,
// thread = channel, per-pixel loads coalesced across lanes.
// ---------------------------------------------------------------------------
__global__ void k_bn_stats(const float* __restrict__ h, float* __restrict__ mean,
                           float* __restrict__ var, int P, int C) {
  int c = threadIdx.x;
  float s = 0.f, q = 0.f;
  for (int p = 0; p < P; ++p) {
    float v = h[(size_t)p * C + c];
    s += v; q += v * v;
  }
  float inv = 1.f / (float)P;
  float mu = s * inv;
  mean[c] = mu;
  var[c] = q * inv - mu * mu;
}

__global__ void k_bn_apply(const float* __restrict__ h, const float* __restrict__ gamma,
                           const float* __restrict__ beta, const float* __restrict__ mean,
                           const float* __restrict__ var, _Float16* __restrict__ out,
                           long total, int C) {
  long i = (long)blockIdx.x * blockDim.x + threadIdx.x;
  if (i >= total) return;
  int c = (int)(i % C);
  float v = (h[i] - mean[c]) * rsqrtf(var[c] + BN_EPS) * gamma[c] + beta[c];
  v = (v >= 0.f) ? v : LRELU_SLOPE * v;
  out[i] = (_Float16)v;
}

// ---------------------------------------------------------------------------
// 1x1 epilogue, unpadded NHWC f16 in, NCHW f32 out.
// ---------------------------------------------------------------------------
__global__ void k_epilogue(const _Float16* __restrict__ h, const float* __restrict__ wep,
                           const float* __restrict__ bep, float* __restrict__ out,
                           int N, int C, int HW) {
  long i = (long)blockIdx.x * blockDim.x + threadIdx.x;  // (n, hw)
  if (i >= (long)N * HW) return;
  int n = (int)(i / HW);
  int k = (int)(i - (long)n * HW);
  const v8h* hp = (const v8h*)(h + i * C);
  float a0 = 0.f, a1 = 0.f, a2 = 0.f;
#pragma unroll
  for (int g = 0; g < 8; ++g) {
    v8h hv = hp[g];
#pragma unroll
    for (int r = 0; r < 8; ++r) {
      float v = (float)hv[r];
      int c = g * 8 + r;
      a0 += v * wep[c];
      a1 += v * wep[C + c];
      a2 += v * wep[2 * C + c];
    }
  }
  size_t o = ((size_t)n * 3) * HW + k;
  out[o]          = a0 + bep[0];
  out[o + HW]     = a1 + bep[1];
  out[o + 2 * HW] = a2 + bep[2];
}

// ---------------------------------------------------------------------------
// Host orchestration
// ---------------------------------------------------------------------------
static inline size_t align256(size_t x) { return (x + 255) & ~(size_t)255; }

extern "C" void kernel_launch(void* const* d_in, const int* in_sizes, int n_in,
                              void* d_out, int out_size, void* d_ws, size_t ws_size,
                              hipStream_t stream) {
  (void)in_sizes; (void)n_in; (void)out_size; (void)ws_size;

  const float* x = (const float*)d_in[0];
  const float *w_d[5], *b_d[5], *w_u[5], *b_u[5];
  for (int i = 0; i < 5; ++i) { w_d[i] = (const float*)d_in[1 + 2 * i];  b_d[i] = (const float*)d_in[2 + 2 * i]; }
  const float* w_in  = (const float*)d_in[11];
  const float* b_in  = (const float*)d_in[12];
  const float* gamma = (const float*)d_in[13];
  const float* beta  = (const float*)d_in[14];
  for (int i = 0; i < 5; ++i) { w_u[i] = (const float*)d_in[15 + 2 * i]; b_u[i] = (const float*)d_in[16 + 2 * i]; }
  const float* w_ep = (const float*)d_in[25];
  const float* b_ep = (const float*)d_in[26];
  float* out = (float*)d_out;

  char* ws = (char*)d_ws;
  size_t cur = 0;
  auto alloc = [&](size_t bytes) -> void* {
    void* p = ws + cur;
    cur = align256(cur + bytes);
    return p;
  };

  // max padded activation: (2, 514, 514, 64) f16
  const size_t BUF = (size_t)2 * 514 * 514 * 64 * sizeof(_Float16);
  _Float16* bufA = (_Float16*)alloc(BUF);
  _Float16* bufB = (_Float16*)alloc(BUF);
  float* inner   = (float*)alloc((size_t)2 * 16 * 16 * 1024 * sizeof(float));
  float* bn_mean = (float*)alloc(1024 * sizeof(float));
  float* bn_var  = (float*)alloc(1024 * sizeof(float));

  auto pack = [&](const float* w, int Cout, int Cin) -> _Float16* {
    int K = Cin * 9;
    int nKt = (K + 31) / 32;
    int nMt = (Cout + 15) / 16;
    int total = nMt * nKt * 512;
    _Float16* wp = (_Float16*)alloc((size_t)total * sizeof(_Float16));
    k_pack_w<<<(total + 255) / 256, 256, 0, stream>>>(w, wp, Cout, Cin, K, nKt, total);
    return wp;
  };

  const int cd_in[5]  = {3, 64, 128, 256, 512};
  const int cd_out[5] = {64, 128, 256, 512, 1024};
  const int cu_in[5]  = {1024, 512, 256, 128, 64};
  const int cu_out[5] = {512, 256, 128, 64, 64};

  _Float16 *p_d[5], *p_u[5];
  for (int i = 0; i < 5; ++i) p_d[i] = pack(w_d[i], cd_out[i], cd_in[i]);
  _Float16* p_in = pack(w_in, 1024, 1024);
  for (int i = 0; i < 5; ++i) p_u[i] = pack(w_u[i], cu_out[i], cu_in[i]);

  auto zero_border = [&](_Float16* buf, int Hp, int Wp, int C) {
    long total = (long)2 * (2 * Wp + 2 * (Hp - 2)) * C;
    k_zero_border<<<(unsigned)((total + 255) / 256), 256, 0, stream>>>(buf, 2, Hp, Wp, C);
  };
  auto conv = [&](const _Float16* in, const _Float16* wp, const float* bias,
                  _Float16* oh, float* of,
                  int N, int Cin, int H, int W, int Cout, int Ho, int Wo,
                  int stride, int lrelu, int opad) {
    long P = (long)N * Ho * Wo;  // divisible by 128 for all layers here
    if (Cin % 32 == 0) {
      dim3 g((unsigned)(Cout / 16), (unsigned)(P / 128));
      k_conv3x3_nhwc<<<g, 128, 0, stream>>>(in, wp, bias, oh, of,
                                            N, Cin, H, W, Cout, Ho, Wo, stride, lrelu, opad);
    } else {
      dim3 g((unsigned)(Cout / 16), (unsigned)(P / 64));
      k_conv3x3_nhwc_small<<<g, 128, 0, stream>>>(in, wp, bias, oh,
                                                  N, Cin, H, W, Cout, Ho, Wo, stride, lrelu, opad);
    }
  };
  auto up = [&](const _Float16* in, _Float16* o, int C, int H, int W) {
    zero_border(o, 2 * H + 2, 2 * W + 2, C);
    long total = (long)2 * 4 * H * W * C;
    k_up2<<<(unsigned)((total + 255) / 256), 256, 0, stream>>>(in, o, 2, C, H, W);
  };

  // input NCHW f32 -> padded NHWC f16
  zero_border(bufA, 514, 514, 3);
  {
    long n = (long)2 * 512 * 512 * 3;
    k_cvt_in<<<(unsigned)((n + 255) / 256), 256, 0, stream>>>(x, bufA, 2, 512, 512);
  }

  // encoder (fused stride-2), padded outputs
  int H = 512;
  _Float16* src = bufA;
  _Float16* dst = bufB;
  for (int i = 0; i < 5; ++i) {
    zero_border(dst, H / 2 + 2, H / 2 + 2, cd_out[i]);
    conv(src, p_d[i], b_d[i], dst, nullptr,
         2, cd_in[i], H, H, cd_out[i], H / 2, H / 2, 2, 1, /*opad=*/1);
    H /= 2;
    _Float16* t = src; src = dst; dst = t;
  }
  // H == 16, src = padded (2,18,18,1024) f16

  // innermost conv (unpadded f32 out) + BN(batch stats) + lrelu -> unpadded f16
  conv(src, p_in, b_in, nullptr, inner, 2, 1024, 16, 16, 1024, 16, 16, 1, 0, /*opad=*/0);
  k_bn_stats<<<1, 1024, 0, stream>>>(inner, bn_mean, bn_var, 2 * 16 * 16, 1024);
  {
    long n = (long)2 * 16 * 16 * 1024;
    k_bn_apply<<<(unsigned)((n + 255) / 256), 256, 0, stream>>>(
        inner, gamma, beta, bn_mean, bn_var, src, n, 1024);
  }

  // decoder: padded upsample, unpadded conv outputs
  for (int i = 0; i < 5; ++i) {
    up(src, dst, cu_in[i], H, H);
    H *= 2;
    _Float16* t = src; src = dst; dst = t;
    conv(src, p_u[i], b_u[i], dst, nullptr,
         2, cu_in[i], H, H, cu_out[i], H, H, 1, 1, /*opad=*/0);
    t = src; src = dst; dst = t;
  }
  // H == 512, src = unpadded (2,512,512,64) f16

  // 1x1 epilogue -> NCHW f32 output
  {
    long n = (long)2 * 512 * 512;
    k_epilogue<<<(unsigned)((n + 255) / 256), 256, 0, stream>>>(
        src, w_ep, b_ep, out, 2, 64, 512 * 512);
  }
}
